// SlidingMultiHeadAttention_21715354649908
// MI455X (gfx1250) — matrix operand order repaired and checked by reading, not compiled, not run on hardware
//
#include <hip/hip_runtime.h>

typedef _Float16 f16;
typedef __attribute__((ext_vector_type(2)))  _Float16 v2h;
typedef __attribute__((ext_vector_type(4)))  _Float16 v4h;
typedef __attribute__((ext_vector_type(8)))  _Float16 v8h;
typedef __attribute__((ext_vector_type(16))) _Float16 v16h;
typedef __attribute__((ext_vector_type(8)))  float    v8f;
typedef __attribute__((ext_vector_type(4)))  unsigned int v4u;
typedef __attribute__((ext_vector_type(8)))  int      v8i;
typedef __attribute__((ext_vector_type(4)))  int      v4i;

#define C_DIM   512
#define ROWS    64          // tokens per block (8 windows)
#define XS_STR  520         // padded row stride (halfs) to spread LDS banks
#define VT_TOK  80          // 64 tokens + 16 zero-pad for K=32 WMMA
#define NWIN    512         // windows per batch row (L/S)
#define QSCALE  0.044194173824159216f   // 1/sqrt(512)

struct Smem {
  f16 xs[ROWS][XS_STR];     // x tile (f16); reused as Y (attention output)
  f16 Qs[ROWS][XS_STR];     // Q (pre-scaled, +bias), row-major [token][dim]
  f16 Ks[ROWS][XS_STR];     // K (+bias), row-major [token][dim]
  f16 Vt[C_DIM][VT_TOK];    // V transposed [dim][token]; tokens 64..79 = 0
  f16 att[8][16][16];       // per-wave softmax scratch
  float biasT[15][8];       // relative-position bias table
};
// TDM f32 staging tile (64 x 512 f32 = 128 KB) aliases Qs+Ks (133 KB),
// which are dead until the phase-1 epilogue.

__device__ __forceinline__ v16h cat16(v8h lo, v8h hi) {
  return __builtin_shufflevector(lo, hi, 0,1,2,3,4,5,6,7,8,9,10,11,12,13,14,15);
}
__device__ __forceinline__ v8f zero8f() {
  v8f z;
#pragma unroll
  for (int i = 0; i < 8; ++i) z[i] = 0.0f;
  return z;
}
__device__ __forceinline__ v8h zero8h() {
  v8h z;
#pragma unroll
  for (int i = 0; i < 8; ++i) z[i] = (f16)0.0f;
  return z;
}

// A-fragment (16x32 f16, M x K): lane holds row M=lane%16; halves 0-7 = K[8*(lane/16)+i],
// halves 8-15 = K[16+8*(lane/16)+i]  -> two 16B loads from a row-major [row][K] array.
template <int STRIDE>
__device__ __forceinline__ v16h loadA(const f16* base, int m0, int k0, int lane) {
  const int lr = lane & 15, lhi = lane >> 4;
  const f16* p = base + (m0 + lr) * STRIDE + k0 + 8 * lhi;
  v8h lo = *(const v8h*)p;
  v8h hi = *(const v8h*)(p + 16);
  return cat16(lo, hi);
}

// B-fragment (32x16 f16, K x N): lane holds col N=lane%16; halves 0-7 = K[8*(lane/16)+i],
// halves 8-15 = K[16+8*(lane/16)+i] -> two 16B loads from a [col][K] array.
template <int STRIDE>
__device__ __forceinline__ v16h loadB(const f16* base, int n0, int k0, int lane) {
  const int lr = lane & 15, lhi = lane >> 4;
  const f16* p = base + (size_t)(n0 + lr) * STRIDE + k0 + 8 * lhi;
  v8h lo = *(const v8h*)p;
  v8h hi = *(const v8h*)(p + 16);
  return cat16(lo, hi);
}

__device__ __forceinline__ v8f wmma_f16(v16h a, v16h b, v8f c) {
  return __builtin_amdgcn_wmma_f32_16x16x32_f16(false, a, false, b, (short)0, c, false, false);
}

// ---------------------------------------------------------------------------
// Kernel 1: convert W{q,k,v,p} (f32, [K][N]) -> f16 transposed Wt[n][k]
// ---------------------------------------------------------------------------
__global__ void prep_weights_f16t(const float* __restrict__ wq, const float* __restrict__ wk,
                                  const float* __restrict__ wv, const float* __restrict__ wp,
                                  f16* __restrict__ wt) {
  int idx = blockIdx.x * 256 + threadIdx.x;       // 4 * 512 * 512 elements
  int mat = idx >> 18;
  int e   = idx & 262143;
  int n   = e >> 9;
  int k   = e & 511;
  const float* src = (mat == 0) ? wq : (mat == 1) ? wk : (mat == 2) ? wv : wp;
  wt[(size_t)mat * 262144 + n * 512 + k] = (f16)src[k * 512 + n];
}

// ---------------------------------------------------------------------------
// Kernel 2: fused roll -> QKV -> windowed attention -> projection -> unroll
// grid = B * (L/ROWS) = 32 * 64 = 2048 blocks of 256 threads (8 waves)
// ---------------------------------------------------------------------------
__global__ __launch_bounds__(256) void swin_attn_fused(
    const float* __restrict__ x,
    const f16* __restrict__ wqt, const f16* __restrict__ wkt,
    const f16* __restrict__ wvt, const f16* __restrict__ wpt,
    const float* __restrict__ bq, const float* __restrict__ bk,
    const float* __restrict__ bv, const float* __restrict__ bp,
    const float* __restrict__ bias_table,
    float* __restrict__ out) {
  extern __shared__ __align__(16) char smraw[];
  Smem& sm = *reinterpret_cast<Smem*>(smraw);

  const int tid  = threadIdx.x;
  const int lane = tid & 31;
  const int w    = tid >> 5;                 // wave id 0..7
  const int lr   = lane & 15;
  const int lhi  = lane >> 4;
  const int bb   = blockIdx.x >> 6;          // batch
  const int rloc = (blockIdx.x & 63) << 6;   // first rolled token of tile

  const float* xb = x + (size_t)bb * 4096 * 512;

  // ---------------- Phase 0: stage x tile (roll +4) and convert to f16 ------
#if defined(__gfx1250__) && __has_builtin(__builtin_amdgcn_tensor_load_to_lds) && \
    __has_builtin(__builtin_amdgcn_s_wait_tensorcnt)
  float* staging = (float*)&sm.Qs[0][0];     // 64 x 512 f32, aliases Qs/Ks
  if (w == 0) {
    // Tensor DMA: 2D tile, rows of 512 f32, tile start = rolled row 0 of block.
    long row0  = (long)bb * 4096 + rloc + 4;           // flat source row
    long avail = 131072l - row0;
    unsigned tile_rows = (avail >= 64) ? 64u : (unsigned)avail;   // clamp at buffer end
    unsigned long long ga = (unsigned long long)(uintptr_t)(x + (size_t)row0 * 512);
    unsigned lds_off = (unsigned)(uintptr_t)staging;   // low 32 bits = LDS byte offset
    // D# group 0: count=1 | lds_addr | global_addr[56:0] | type=2 (bits 127:126)
    v4u g0 = { 1u, lds_off, (unsigned)(ga & 0xffffffffu),
               (unsigned)(ga >> 32) | 0x80000000u };
    // D# group 1 (ISA 8.4): data_size=2(4B); tensor_dim0=512; tensor_dim1=tile_rows;
    // tile_dim0=512; tile_dim1=tile_rows; tensor_dim0_stride=512.
    unsigned d0 = (2u << 16);
    unsigned d1 = (512u << 16);
    unsigned d2 = (tile_rows & 0xffffu) << 16;
    unsigned d3 = (512u << 16);
    unsigned d4 = (tile_rows & 0xffffu);
    unsigned d5 = 512u;
    v8i g1 = { (int)d0, (int)d1, (int)d2, (int)d3, (int)d4, (int)d5, 0, 0 };
    v4i gz4 = { 0, 0, 0, 0 };
    v8i gz8 = { 0, 0, 0, 0, 0, 0, 0, 0 };
    // 6-arg form (clang-23 / therock-10.0 headers): (g0, g1, g2, g3, g4, cpol)
    __builtin_amdgcn_tensor_load_to_lds(g0, g1, gz4, gz4, gz8, 0);
  }
  // Independent init while the TDM runs: V zero-pad + bias table.
#pragma unroll
  for (int it = 0; it < 32; ++it) {
    int i = it * 256 + tid;                  // 512 * 16
    sm.Vt[i >> 4][64 + (i & 15)] = (f16)0.0f;
  }
  if (tid < 120) (&sm.biasT[0][0])[tid] = bias_table[tid];
  if (w == 0) __builtin_amdgcn_s_wait_tensorcnt(0);
  __syncthreads();
  // Convert f32 staging -> f16 xs; wrapped rows (roll past L) come from global.
#pragma unroll
  for (int it = 0; it < 32; ++it) {
    int i   = it * 256 + tid;                // float4 index, 64*128 total
    int row = i >> 7;
    int c4  = i & 127;
    int srcrow = rloc + 4 + row;
    float4 v;
    if (srcrow < 4096) v = *(const float4*)(staging + (size_t)row * 512 + c4 * 4);
    else               v = *(const float4*)(xb + (size_t)(srcrow - 4096) * 512 + c4 * 4);
    v4h h; h[0] = (f16)v.x; h[1] = (f16)v.y; h[2] = (f16)v.z; h[3] = (f16)v.w;
    *(v4h*)&sm.xs[row][c4 * 4] = h;
  }
  __syncthreads();
#else
  // Fallback path (host pass / toolchains without the TDM builtin).
#pragma unroll
  for (int it = 0; it < 32; ++it) {
    int i   = it * 256 + tid;
    int row = i >> 7;
    int c4  = i & 127;
    int src = (rloc + row + 4) & 4095;
    float4 v = *(const float4*)(xb + (size_t)src * 512 + c4 * 4);
    v4h h; h[0] = (f16)v.x; h[1] = (f16)v.y; h[2] = (f16)v.z; h[3] = (f16)v.w;
    *(v4h*)&sm.xs[row][c4 * 4] = h;
  }
#pragma unroll
  for (int it = 0; it < 32; ++it) {
    int i = it * 256 + tid;
    sm.Vt[i >> 4][64 + (i & 15)] = (f16)0.0f;
  }
  if (tid < 120) (&sm.biasT[0][0])[tid] = bias_table[tid];
  __syncthreads();
#endif

  // ---------------- Phase 1: QKV = xs @ W{q,k,v}^T(streamed) + b ------------
  // Wave w owns N-tiles 4w..4w+3; 4 M-tiles x 4 N-tiles of f32 accumulators.
  {
    const f16*   wts[3] = {wqt, wkt, wvt};
    const float* bss[3] = {bq, bk, bv};
#pragma unroll
    for (int mat = 0; mat < 3; ++mat) {
      const f16* wt = wts[mat];
      v8f acc[4][4];
#pragma unroll
      for (int m = 0; m < 4; ++m)
#pragma unroll
        for (int j = 0; j < 4; ++j) acc[m][j] = zero8f();

      for (int kt = 0; kt < 16; ++kt) {
        int k0 = kt * 32;
        if (kt + 1 < 16)
          __builtin_prefetch((const void*)(wt + (size_t)((w * 4) * 16 + lr) * 512 + k0 + 32), 0, 3);
        v16h a[4], bf[4];
#pragma unroll
        for (int m = 0; m < 4; ++m) a[m] = loadA<XS_STR>(&sm.xs[0][0], m * 16, k0, lane);
#pragma unroll
        for (int j = 0; j < 4; ++j) bf[j] = loadB<512>(wt, (w * 4 + j) * 16, k0, lane);
#pragma unroll
        for (int m = 0; m < 4; ++m)
#pragma unroll
          for (int j = 0; j < 4; ++j) acc[m][j] = wmma_f16(a[m], bf[j], acc[m][j]);
      }
      // epilogue: +bias; Q scaled by 1/sqrt(C); Q/K row-major, V transposed
#pragma unroll
      for (int j = 0; j < 4; ++j) {
        int n = (w * 4 + j) * 16 + lr;
        float bias = bss[mat][n];
#pragma unroll
        for (int m = 0; m < 4; ++m) {
          if (mat == 2) {
            // V: packed half2 stores, consecutive rows = consecutive tokens
#pragma unroll
            for (int rr = 0; rr < 4; ++rr) {
              v2h pk;
              pk[0] = (f16)(acc[m][j][2 * rr]     + bias);
              pk[1] = (f16)(acc[m][j][2 * rr + 1] + bias);
              *(v2h*)&sm.Vt[n][m * 16 + 2 * rr + 8 * lhi] = pk;
            }
          } else {
#pragma unroll
            for (int r = 0; r < 8; ++r) {
              float val = acc[m][j][r] + bias;
              int tok = m * 16 + r + 8 * lhi;
              if (mat == 0) sm.Qs[tok][n] = (f16)(val * QSCALE);
              else          sm.Ks[tok][n] = (f16)val;
            }
          }
        }
      }
    }
  }
  __syncthreads();

  // ---------------- Phase 2: windowed attention (2 windows / 16x16 tile) ----
  // 32 units = 4 window-pairs x 8 heads; wave w handles units 4w..4w+3.
#pragma unroll
  for (int uu = 0; uu < 4; ++uu) {
    int u  = w * 4 + uu;
    int wp = u >> 3;                 // window pair 0..3
    int h  = u & 7;                  // head

    v8f e = zero8f();
#pragma unroll
    for (int kk = 0; kk < 64; kk += 32) {
      v16h aq  = loadA<XS_STR>(&sm.Qs[0][0], wp * 16, h * 64 + kk, lane);
      v16h bk_ = loadB<XS_STR>(&sm.Ks[0][0], wp * 16, h * 64 + kk, lane);
      e = wmma_f16(aq, bk_, e);
    }

    // element (m=r+8*lhi, n=lr): valid iff row-window == col-window
    int  kj    = lane & 7;
    bool valid = (lhi == (lr >> 3));
    int  wglob = (rloc >> 3) + wp * 2 + lhi;      // global window of the row
    bool lastw = (wglob == NWIN - 1);

#pragma unroll
    for (int r = 0; r < 8; ++r) {
      float val = e[r] + sm.biasT[kj - r + 7][h];
      if (lastw && ((r < 4) != (kj < 4))) val -= 100.0f;
      // softmax across the 8 in-window key columns (8-lane groups)
      float mx = val;
      mx = fmaxf(mx, __shfl_xor(mx, 1, 8));
      mx = fmaxf(mx, __shfl_xor(mx, 2, 8));
      mx = fmaxf(mx, __shfl_xor(mx, 4, 8));
      float p = __expf(val - mx);
      float s = p;
      s += __shfl_xor(s, 1, 8);
      s += __shfl_xor(s, 2, 8);
      s += __shfl_xor(s, 4, 8);
      float a = valid ? (p / s) : 0.0f;           // zero cross-window block
      sm.att[w][r + 8 * lhi][lr] = (f16)a;
    }

    // AV: out = att(16x16, zero-padded to K=32) @ V(16x64); Y overwrites xs
    v8h alo = *(const v8h*)&sm.att[w][lr][8 * lhi];
    v16h aatt = cat16(alo, zero8h());
#pragma unroll
    for (int j = 0; j < 4; ++j) {
      v16h bv_ = loadB<VT_TOK>(&sm.Vt[0][0], h * 64 + j * 16, wp * 16, lane);
      v8f o = wmma_f16(aatt, bv_, zero8f());
#pragma unroll
      for (int r = 0; r < 8; ++r) {
        int tok = wp * 16 + r + 8 * lhi;
        sm.xs[tok][h * 64 + j * 16 + lr] = (f16)o[r];
      }
    }
  }
  __syncthreads();

  // ---------------- Phase 3: projection Y @ Wp^T + bp, inverse roll ---------
  {
    v8f acc[4][4];
#pragma unroll
    for (int m = 0; m < 4; ++m)
#pragma unroll
      for (int j = 0; j < 4; ++j) acc[m][j] = zero8f();

    for (int kt = 0; kt < 16; ++kt) {
      int k0 = kt * 32;
      if (kt + 1 < 16)
        __builtin_prefetch((const void*)(wpt + (size_t)((w * 4) * 16 + lr) * 512 + k0 + 32), 0, 3);
      v16h a[4], bf[4];
#pragma unroll
      for (int m = 0; m < 4; ++m) a[m] = loadA<XS_STR>(&sm.xs[0][0], m * 16, k0, lane);
#pragma unroll
      for (int j = 0; j < 4; ++j) bf[j] = loadB<512>(wpt, (w * 4 + j) * 16, k0, lane);
#pragma unroll
      for (int m = 0; m < 4; ++m)
#pragma unroll
        for (int j = 0; j < 4; ++j) acc[m][j] = wmma_f16(a[m], bf[j], acc[m][j]);
    }

    float* ob = out + (size_t)bb * 4096 * 512;
#pragma unroll
    for (int j = 0; j < 4; ++j) {
      int n = (w * 4 + j) * 16 + lr;
      float bias = bp[n];
#pragma unroll
      for (int m = 0; m < 4; ++m) {
#pragma unroll
        for (int r = 0; r < 8; ++r) {
          int row  = rloc + m * 16 + r + 8 * lhi;     // rolled index
          int orow = (row + 4) & 4095;                // roll back (+S/2)
          ob[(size_t)orow * 512 + n] = acc[m][j][r] + bias;
        }
      }
    }
  }
}

// ---------------------------------------------------------------------------
extern "C" void kernel_launch(void* const* d_in, const int* in_sizes, int n_in,
                              void* d_out, int out_size, void* d_ws, size_t ws_size,
                              hipStream_t stream) {
  (void)in_sizes; (void)n_in; (void)out_size; (void)ws_size;
  const float* x  = (const float*)d_in[0];
  const float* Wq = (const float*)d_in[1];
  const float* bq = (const float*)d_in[2];
  const float* Wk = (const float*)d_in[3];
  const float* bk = (const float*)d_in[4];
  const float* Wv = (const float*)d_in[5];
  const float* bv = (const float*)d_in[6];
  const float* Wp = (const float*)d_in[7];
  const float* bp = (const float*)d_in[8];
  const float* bt = (const float*)d_in[9];

  f16* wt = (f16*)d_ws;                 // 4 x 512x512 f16 transposed weights (2 MB)
  prep_weights_f16t<<<4096, 256, 0, stream>>>(Wq, Wk, Wv, Wp, wt);

  const size_t smem = sizeof(Smem);     // ~286 KB (WGP has 320 KB LDS)
  (void)hipFuncSetAttribute(reinterpret_cast<const void*>(swin_attn_fused),
                            hipFuncAttributeMaxDynamicSharedMemorySize, (int)smem);
  swin_attn_fused<<<2048, 256, smem, stream>>>(
      x, wt, wt + 262144, wt + 524288, wt + 786432,
      bq, bk, bv, bp, bt, (float*)d_out);
}
